// sensitivity_prediction_11716670783533
// MI455X (gfx1250) — compile-verified, matrix operand order checked by reading, not run on hardware
//
#include <hip/hip_runtime.h>

#define NE        8192
#define BATCH     64
#define NNZ       262144
#define NEG_SLOPE 0.001f
#define KSPLIT    8           // k-chunks per output tile (parallelism for HBM streaming)
#define WAVES_PER_BLOCK 8     // 256 threads = 8 wave32

typedef __attribute__((ext_vector_type(16))) __bf16 v16bf;
typedef __attribute__((ext_vector_type(8)))  float  v8f;
typedef __attribute__((ext_vector_type(4)))  float  v4f;

// Load one wave32 bf16 fragment (16 rows x 32 k, this lane's 16 elements) from an
// f32 row-major source. Lane l: row = l&15; k-halves at kOff and kOff+16 where
// kOff = (l>=16)?8:0 — matches the CDNA5 16-bit A/B VGPR striping.
// Caller passes p already offset by (row*NE + kBase + kOff).
__device__ __forceinline__ v16bf load_frag_bf16(const float* __restrict__ p) {
    v4f a0 = *(const v4f*)(p + 0);
    v4f a1 = *(const v4f*)(p + 4);
    v4f a2 = *(const v4f*)(p + 16);
    v4f a3 = *(const v4f*)(p + 20);
    v16bf r;
#pragma unroll
    for (int i = 0; i < 4; ++i) {
        r[i]      = (__bf16)a0[i];
        r[i + 4]  = (__bf16)a1[i];
        r[i + 8]  = (__bf16)a2[i];
        r[i + 12] = (__bf16)a3[i];
    }
    return r;
}

// --- Kernel 1: zero the f32 accumulators in workspace (hT + outT) -----------
__global__ __launch_bounds__(256) void zero_ws_kernel(float* __restrict__ p, int n) {
    int t = blockIdx.x * 256 + threadIdx.x;
    if (t < n) p[t] = 0.0f;
}

// --- Kernel 2: hT[j][b] = sum_k x[b][k] * W[j][k]  (WMMA bf16, k-split) -----
// One wave handles one N-tile (16 j columns) x all 4 M-tiles (64 batch rows)
// for one k-chunk, so each W element is loaded from HBM exactly once.
__global__ __launch_bounds__(256) void gemm_xwt_kernel(const float* __restrict__ x,
                                                       const float* __restrict__ W,
                                                       float* __restrict__ hT) {
    const int lane  = threadIdx.x & 31;
    const int wave  = threadIdx.x >> 5;
    const int nTile = blockIdx.x * WAVES_PER_BLOCK + wave;   // 0..511
    const int nBase = nTile * 16;
    const int r     = lane & 15;
    const int kOff  = (lane >> 4) * 8;

    const int kChunk = NE / KSPLIT;                 // 1024
    const int kBase0 = blockIdx.y * kChunk;

    const float* wrow = W + (size_t)(nBase + r) * NE + kBase0 + kOff;  // B: W[j][k]
    const float* x0   = x + (size_t)(r)            * NE + kBase0 + kOff;
    const float* x1   = x0 + (size_t)16 * NE;
    const float* x2   = x1 + (size_t)16 * NE;
    const float* x3   = x2 + (size_t)16 * NE;

    v8f acc0 = {}, acc1 = {}, acc2 = {}, acc3 = {};

    for (int k = 0; k < kChunk; k += 32) {
        // prefetch next W segment of this row (gfx1250 global_prefetch_b8)
        __builtin_prefetch(wrow + k + 32, 0, 1);

        v16bf bf = load_frag_bf16(wrow + k);
        v16bf a0 = load_frag_bf16(x0 + k);
        v16bf a1 = load_frag_bf16(x1 + k);
        v16bf a2 = load_frag_bf16(x2 + k);
        v16bf a3 = load_frag_bf16(x3 + k);

        acc0 = __builtin_amdgcn_wmma_f32_16x16x32_bf16(false, a0, false, bf,
                                                       (short)0, acc0, false, false);
        acc1 = __builtin_amdgcn_wmma_f32_16x16x32_bf16(false, a1, false, bf,
                                                       (short)0, acc1, false, false);
        acc2 = __builtin_amdgcn_wmma_f32_16x16x32_bf16(false, a2, false, bf,
                                                       (short)0, acc2, false, false);
        acc3 = __builtin_amdgcn_wmma_f32_16x16x32_bf16(false, a3, false, bf,
                                                       (short)0, acc3, false, false);
    }

    // C/D layout: elem e of lane l holds (m = e + (l>>4)*8, n = l&15).
    // Accumulate transposed: hT[(nBase+n)*64 + m]  (k-split -> atomic f32 add, L2).
    float* col = hT + (size_t)(nBase + r) * BATCH + (lane >> 4) * 8;
#pragma unroll
    for (int e = 0; e < 8; ++e) {
        atomicAdd(&col[ 0 + e], acc0[e]);
        atomicAdd(&col[16 + e], acc1[e]);
        atomicAdd(&col[32 + e], acc2[e]);
        atomicAdd(&col[48 + e], acc3[e]);
    }
}

// --- Kernel 3: SpMM  outT[j][b] += w * hT[i][b] -----------------------------
// 64 consecutive threads share one nonzero -> fully contiguous 256B read/update.
__global__ __launch_bounds__(256) void spmm_kernel(const long long* __restrict__ idx,
                                                   const float* __restrict__ wts,
                                                   const float* __restrict__ hT,
                                                   float* __restrict__ outT) {
    unsigned t  = blockIdx.x * 256u + threadIdx.x;
    unsigned nz = t >> 6;
    unsigned b  = t & 63u;
    int   i = (int)idx[2 * (size_t)nz + 0];
    int   j = (int)idx[2 * (size_t)nz + 1];
    float w = wts[nz];
    atomicAdd(&outT[(size_t)j * BATCH + b], w * hT[(size_t)i * BATCH + b]);
}

// --- Kernel 4: transpose back + LeakyReLU -----------------------------------
__global__ __launch_bounds__(256) void finalize_kernel(const float* __restrict__ outT,
                                                       float* __restrict__ out) {
    int t = blockIdx.x * 256 + threadIdx.x;   // t = b*NE + j
    int b = t >> 13;                          // NE = 2^13
    int j = t & (NE - 1);
    float v = outT[(size_t)j * BATCH + b];
    out[t] = (v >= 0.0f) ? v : NEG_SLOPE * v;
}

extern "C" void kernel_launch(void* const* d_in, const int* in_sizes, int n_in,
                              void* d_out, int out_size, void* d_ws, size_t ws_size,
                              hipStream_t stream) {
    const float*     x   = (const float*)d_in[0];      // (64, 8192) f32
    const float*     W   = (const float*)d_in[1];      // (8192, 8192) f32
    const long long* idx = (const long long*)d_in[2];  // (262144, 2) i64
    const float*     sw  = (const float*)d_in[3];      // (262144,) f32
    float*           out = (float*)d_out;              // (64, 8192) f32

    float* hT   = (float*)d_ws;                        // (NE, BATCH) f32 = 2 MB
    float* outT = hT + (size_t)NE * BATCH;             // (NE, BATCH) f32 = 2 MB

    // 1) zero both accumulators (k-split GEMM and SpMM accumulate via atomics)
    {
        int n = 2 * NE * BATCH;
        zero_ws_kernel<<<(n + 255) / 256, 256, 0, stream>>>(hT, n);
    }
    // 2) hT = (x @ W^T)^T  — 64 blocks x 8 k-splits, 256 threads (8 waves)
    {
        dim3 grid(NE / 16 / WAVES_PER_BLOCK, KSPLIT);  // (64, 8)
        gemm_xwt_kernel<<<grid, 256, 0, stream>>>(x, W, hT);
    }
    // 3) outT += S^T-gather: one nonzero per 64 threads
    {
        long long total = (long long)NNZ * BATCH;      // 16.8M threads
        spmm_kernel<<<(unsigned)(total / 256), 256, 0, stream>>>(idx, sw, hT, outT);
    }
    // 4) transpose + LeakyReLU into d_out
    {
        int n = BATCH * NE;
        finalize_kernel<<<n / 256, 256, 0, stream>>>(outT, out);
    }
}
